// DeltaNet_31877247271562
// MI455X (gfx1250) — matrix operand order repaired and checked by reading, not compiled, not run on hardware
//
#include <hip/hip_runtime.h>
#include <hip/hip_bf16.h>

typedef __attribute__((ext_vector_type(16))) __bf16 v16bf;
typedef __attribute__((ext_vector_type(8)))  float  v8f;

#define CHK 32
#define DIM 64

// ---- WMMA fragment helpers (CDNA5 wave32, 16x16x32 bf16) -------------------
// A fragment (16x32, MxK): lane m = lane&15; elem j<8 -> K=kb+j, j>=8 -> K=kb+16+(j-8),
// kb = (lane>=16 ? 8 : 0).  TRANS=true means source is stored src[k][m].
template <typename T, bool TRANS>
__device__ __forceinline__ v16bf fragA(const T* src, int ld, int kofs, int lane, float scale) {
    const int m  = lane & 15;
    const int kb = ((lane >> 4) << 3) + kofs;
    v16bf f;
#pragma unroll
    for (int j = 0; j < 8; ++j) {
        const int k0 = kb + j, k1 = kb + 16 + j;
        float lo = TRANS ? (float)src[k0 * ld + m] : (float)src[m * ld + k0];
        float hi = TRANS ? (float)src[k1 * ld + m] : (float)src[m * ld + k1];
        f[j]     = (__bf16)(lo * scale);
        f[j + 8] = (__bf16)(hi * scale);
    }
    return f;
}

// B fragment (32x16, KxN): lane n = lane&15; elem e<8 -> K=kb+e, e>=8 -> K=kb+16+(e-8), kb = (lane>=16 ? 8:0) + kofs.
// TRANS=true means source is stored src[n][k] (contiguous along k -> vector ds loads).
template <typename T, bool TRANS>
__device__ __forceinline__ v16bf fragB(const T* src, int ld, int kofs, int lane) {
    const int n  = lane & 15;
    const int kb = ((lane >> 4) << 3) + kofs;
    v16bf f;
#pragma unroll
    for (int e = 0; e < 16; ++e) {
        const int k = kb + (e < 8 ? e : e + 8);
        f[e] = (__bf16)(TRANS ? (float)src[n * ld + k] : (float)src[k * ld + n]);
    }
    return f;
}

// C/D fragment (16x16 f32): lane n = lane&15; vgpr r -> row M = r + (lane>=16 ? 8:0)
__device__ __forceinline__ v8f ldC(const float* src, int ld, int lane) {
    const int n = lane & 15, mb = (lane >> 4) * 8;
    v8f c;
#pragma unroll
    for (int r = 0; r < 8; ++r) c[r] = src[(mb + r) * ld + n];
    return c;
}
__device__ __forceinline__ void stC(float* dst, int ld, int lane, v8f c) {
    const int n = lane & 15, mb = (lane >> 4) * 8;
#pragma unroll
    for (int r = 0; r < 8; ++r) dst[(mb + r) * ld + n] = c[r];
}

__device__ __forceinline__ v8f wmma_bf16(v16bf a, v16bf b, v8f c) {
    return __builtin_amdgcn_wmma_f32_16x16x32_bf16(false, a, false, b, (short)0, c,
                                                   false, false);
}

// ---- DeltaNet chunkwise kernel: one workgroup (8 waves) per (b,h) ----------
__global__ __launch_bounds__(256) void deltanet_chunk_kernel(
    const float* __restrict__ q, const float* __restrict__ k,
    const float* __restrict__ v, const float* __restrict__ beta,
    float* __restrict__ o, float* __restrict__ Sout, int L) {
    __shared__ float  sS[DIM][DIM];     // running state, fp32 master
    __shared__ __bf16 sSt[DIM][DIM];    // S^T as bf16: sSt[n][k] = S[k][n]  (B operand)
    __shared__ float  sU[CHK][DIM];     // u, then u_i, fp32 (WMMA C operand)
    __shared__ float  sM[CHK][CHK];     // M = I + strict_lower(A); later attn
    __shared__ float  sT[CHK][CHK];     // T = M^{-1}, fp32
    __shared__ __bf16 sQ[CHK][DIM];     // l2norm(q)              (A operand)
    __shared__ __bf16 sK[CHK][DIM];     // l2norm(k)              (A^T / B^T operand)
    __shared__ __bf16 sVt[DIM][CHK];    // (v*beta)^T             (B operand)
    __shared__ __bf16 sKB[CHK][DIM];    // l2norm(k)*beta         (A + B operand)
    __shared__ __bf16 sW[CHK][DIM];     // w = T@k_beta; overlaid by u_i^T after stage 5

    __bf16 (*sUt)[CHK] = reinterpret_cast<__bf16(*)[CHK]>(&sW[0][0]);  // u_i^T [64][32]

    const int bh   = blockIdx.x;
    const int t    = threadIdx.x;
    const int lane = t & 31;
    const int wave = t >> 5;
    const int nch  = L / CHK;

    const float* qg0 = q + (size_t)bh * L * DIM;
    const float* kg0 = k + (size_t)bh * L * DIM;
    const float* vg0 = v + (size_t)bh * L * DIM;
    const float* bg0 = beta + (size_t)bh * L;
    float*       og0 = o + (size_t)bh * L * DIM;

    for (int i = t; i < DIM * DIM; i += 256) {
        (&sS[0][0])[i]  = 0.f;
        (&sSt[0][0])[i] = (__bf16)0.f;
    }
    __syncthreads();

    for (int ch = 0; ch < nch; ++ch) {
        const float* qg = qg0 + ch * CHK * DIM;
        const float* kg = kg0 + ch * CHK * DIM;
        const float* vg = vg0 + ch * CHK * DIM;

        // -- stage 1: load + l2norm + beta scaling, staged to LDS as bf16 ----
        {
            const int row = t >> 3, seg = (t & 7) * 8;
            float qr[8], kr[8], vr[8];
            float qs = 0.f, ks = 0.f;
#pragma unroll
            for (int j = 0; j < 8; ++j) {
                qr[j] = qg[row * DIM + seg + j];
                kr[j] = kg[row * DIM + seg + j];
                vr[j] = vg[row * DIM + seg + j];
                qs += qr[j] * qr[j];
                ks += kr[j] * kr[j];
            }
#pragma unroll
            for (int m = 1; m < 8; m <<= 1) {
                qs += __shfl_xor(qs, m, 32);
                ks += __shfl_xor(ks, m, 32);
            }
            const float qsc = rsqrtf(qs + 1e-6f);
            const float ksc = rsqrtf(ks + 1e-6f);
            const float bb  = bg0[ch * CHK + row];
#pragma unroll
            for (int j = 0; j < 8; ++j) {
                sQ[row][seg + j] = (__bf16)(qr[j] * qsc);
                const float kn   = kr[j] * ksc;
                sK[row][seg + j]   = (__bf16)kn;
                sKB[row][seg + j]  = (__bf16)(kn * bb);
                sVt[seg + j][row]  = (__bf16)(vr[j] * bb);
            }
            for (int i = t; i < CHK * CHK; i += 256)
                (&sT[0][0])[i] = ((i >> 5) == (i & 31)) ? 1.f : 0.f;
        }
        __syncthreads();

        // -- stage 2: M = I + strict_lower(k_beta @ k^T)  (waves 0-3) --------
        if (wave < 4) {
            const int ti = wave >> 1, tj = wave & 1;
            v8f acc = {};
#pragma unroll
            for (int kk = 0; kk < DIM; kk += 32) {
                v16bf a = fragA<__bf16, false>(&sKB[ti * 16][0], DIM, kk, lane, 1.f);
                v16bf b = fragB<__bf16, true>(&sK[tj * 16][0], DIM, kk, lane);
                acc = wmma_bf16(a, b, acc);
            }
            const int gn = tj * 16 + (lane & 15);
            const int mb = ti * 16 + ((lane >> 4) * 8);
#pragma unroll
            for (int r = 0; r < 8; ++r) {
                const int gm = mb + r;
                sM[gm][gn] = (gm > gn) ? acc[r] : ((gm == gn) ? 1.f : 0.f);
            }
        } else if (ch + 1 < nch) {
            const int off = (t - 128) * 16;  // 128 threads x 64B cover next chunk
            __builtin_prefetch(qg + CHK * DIM + off, 0, 1);
            __builtin_prefetch(kg + CHK * DIM + off, 0, 1);
            __builtin_prefetch(vg + CHK * DIM + off, 0, 1);
        }
        __syncthreads();

        // -- stage 3: T = M^{-1}, unit-lower forward substitution ------------
        for (int i = 1; i < CHK; ++i) {
            if (t < i) {
                float s = 0.f;
                for (int j = t; j < i; ++j) s += sM[i][j] * sT[j][t];
                sT[i][t] = -s;
            }
            __syncthreads();
        }

        // -- stage 4: u = T@vb (8 tiles), w = T@kb (8), attn (4, into sM) ----
        for (int id = wave; id < 20; id += 8) {
            if (id < 16) {
                const int tt = id & 7, ti = tt >> 2, tj = tt & 3;
                v16bf a = fragA<float, false>(&sT[ti * 16][0], CHK, 0, lane, 1.f);
                v16bf b;
                if (id < 8) b = fragB<__bf16, true>(&sVt[tj * 16][0], CHK, 0, lane);
                else        b = fragB<__bf16, false>(&sKB[0][tj * 16], DIM, 0, lane);
                v8f acc = {};
                acc = wmma_bf16(a, b, acc);
                if (id < 8) {
                    stC(&sU[ti * 16][tj * 16], DIM, lane, acc);
                } else {
                    const int n = lane & 15, mb = (lane >> 4) * 8;
#pragma unroll
                    for (int r = 0; r < 8; ++r)
                        sW[ti * 16 + mb + r][tj * 16 + n] = (__bf16)acc[r];
                }
            } else {
                const int tt = id - 16, ti = tt >> 1, tj = tt & 1;
                v8f acc = {};
#pragma unroll
                for (int kk = 0; kk < DIM; kk += 32) {
                    v16bf a = fragA<__bf16, false>(&sQ[ti * 16][0], DIM, kk, lane, 1.f);
                    v16bf b = fragB<__bf16, true>(&sK[tj * 16][0], DIM, kk, lane);
                    acc = wmma_bf16(a, b, acc);
                }
                const int n = lane & 15, mb = ti * 16 + (lane >> 4) * 8;
#pragma unroll
                for (int r = 0; r < 8; ++r) {
                    const int gm = mb + r, gn = tj * 16 + n;
                    sM[gm][gn] = (gm >= gn) ? acc[r] : 0.f;
                }
            }
        }
        __syncthreads();

        // -- stage 5: u_i = u - w @ S  (negation folded into A fragment) -----
        //    compute first (reads sW), barrier, then store u_i fp32 + u_i^T
        //    bf16 overlaid on sW's storage.
        {
            const int ti = wave >> 2, tj = wave & 3;
            v8f acc = ldC(&sU[ti * 16][tj * 16], DIM, lane);
#pragma unroll
            for (int kk = 0; kk < DIM; kk += 32) {
                v16bf a = fragA<__bf16, false>(&sW[ti * 16][0], DIM, kk, lane, -1.f);
                v16bf b = fragB<__bf16, true>(&sSt[tj * 16][0], DIM, kk, lane);
                acc = wmma_bf16(a, b, acc);
            }
            __syncthreads();  // everyone done reading sW before overlay store
            stC(&sU[ti * 16][tj * 16], DIM, lane, acc);
            const int n = lane & 15, mb = (lane >> 4) * 8;
#pragma unroll
            for (int r = 0; r < 8; ++r)
                sUt[tj * 16 + n][ti * 16 + mb + r] = (__bf16)acc[r];
        }
        __syncthreads();

        // -- stage 6: o = q @ S + attn @ u_i  -> global ----------------------
        {
            const int ti = wave >> 2, tj = wave & 3;
            v8f acc = {};
#pragma unroll
            for (int kk = 0; kk < DIM; kk += 32) {
                v16bf a = fragA<__bf16, false>(&sQ[ti * 16][0], DIM, kk, lane, 1.f);
                v16bf b = fragB<__bf16, true>(&sSt[tj * 16][0], DIM, kk, lane);
                acc = wmma_bf16(a, b, acc);
            }
            v16bf a2 = fragA<float, false>(&sM[ti * 16][0], CHK, 0, lane, 1.f);
            v16bf b2 = fragB<__bf16, true>(&sUt[tj * 16][0], CHK, 0, lane);
            acc = wmma_bf16(a2, b2, acc);
            float* og = og0 + ch * CHK * DIM;
            const int n = lane & 15, mb = (lane >> 4) * 8;
#pragma unroll
            for (int r = 0; r < 8; ++r)
                og[(ti * 16 + mb + r) * DIM + tj * 16 + n] = acc[r];
        }
        __syncthreads();

        // -- stage 7: S += k^T @ u_i (16 tiles, 2/wave); refresh S^T bf16 ----
        for (int id = wave; id < 16; id += 8) {
            const int ti = id >> 2, tj = id & 3;
            v8f acc = ldC(&sS[ti * 16][tj * 16], DIM, lane);
            v16bf a = fragA<__bf16, true>(&sK[0][ti * 16], DIM, 0, lane, 1.f);
            v16bf b = fragB<__bf16, true>(&sUt[tj * 16][0], CHK, 0, lane);
            acc = wmma_bf16(a, b, acc);
            stC(&sS[ti * 16][tj * 16], DIM, lane, acc);
            const int n = lane & 15, mb = (lane >> 4) * 8;
#pragma unroll
            for (int r = 0; r < 8; ++r)
                sSt[tj * 16 + n][ti * 16 + mb + r] = (__bf16)acc[r];
        }
        __syncthreads();
    }

    // final state out
    float* Sg = Sout + (size_t)bh * DIM * DIM;
    for (int i = t; i < DIM * DIM; i += 256) Sg[i] = (&sS[0][0])[i];
}

extern "C" void kernel_launch(void* const* d_in, const int* in_sizes, int n_in,
                              void* d_out, int out_size, void* d_ws, size_t ws_size,
                              hipStream_t stream) {
    (void)n_in; (void)out_size; (void)d_ws; (void)ws_size;
    const float* q    = (const float*)d_in[0];
    const float* k    = (const float*)d_in[1];
    const float* v    = (const float*)d_in[2];
    const float* beta = (const float*)d_in[3];

    const int L  = 4096;                 // reference setup: L = 4096
    const int BH = in_sizes[3] / L;      // beta is [b,h,L] -> b*h streams

    float* o = (float*)d_out;
    float* S = (float*)d_out + (size_t)BH * L * DIM;  // tuple: (o, S) concat

    deltanet_chunk_kernel<<<BH, 256, 0, stream>>>(q, k, v, beta, o, S, L);
}
